// Encoder_70411693851091
// MI455X (gfx1250) — matrix operand order repaired and checked by reading, not compiled, not run on hardware
//
#include <hip/hip_runtime.h>
#include <math.h>

// ---------------------------------------------------------------------------
// CDNA5 (gfx1250) wave32 WMMA types
// ---------------------------------------------------------------------------
typedef __attribute__((ext_vector_type(16))) _Float16 v16h;
typedef __attribute__((ext_vector_type(8)))  _Float16 h8;
typedef __attribute__((ext_vector_type(8)))  float    v8f;

#define ACT_NONE    0
#define ACT_LEAKY   1
#define ACT_RELU    2
#define ACT_SIGMOID 3

// ---------------------------------------------------------------------------
// Weight pack: OIHW f32 -> (Coutp, KH, KW, Cpad) f16, zero-padded in both
// Cout (to mult of 32: two 16-row wave tiles per block) and Cin (to mult of
// 32) so the conv hot loop needs no guards: padded weights are exactly 0.
// ---------------------------------------------------------------------------
__global__ void pack_w_kernel(const float* __restrict__ src, _Float16* __restrict__ dst,
                              int Cout, int Cin, int KH, int KW, int Coutp, int Cpad) {
  int idx = blockIdx.x * blockDim.x + threadIdx.x;
  const int total = Coutp * KH * KW * Cpad;
  if (idx >= total) return;
  int ci = idx % Cpad; int t = idx / Cpad;
  int kw = t % KW; t /= KW;
  int kh = t % KH; t /= KH;
  int co = t;
  float v = (co < Cout && ci < Cin) ? src[((co * Cin + ci) * KH + kh) * KW + kw] : 0.0f;
  dst[idx] = (_Float16)v;
}

// ---------------------------------------------------------------------------
// Activation pack: NCHW f32 -> NHWC f16 slice at channel offset coff with
// row stride Cst; fills Cfill channels (zero beyond logical C).
// ---------------------------------------------------------------------------
__global__ void pack_a_kernel(const float* __restrict__ src, _Float16* __restrict__ dst,
                              int Bn, int C, int H, int W, int Cst, int coff, int Cfill) {
  int idx = blockIdx.x * blockDim.x + threadIdx.x;
  const int total = Bn * H * W * Cfill;
  if (idx >= total) return;
  int cc = idx % Cfill; int t = idx / Cfill;
  int w = t % W; t /= W;
  int h = t % H; t /= H;
  int n = t;
  float v = (cc < C) ? src[((n * C + cc) * H + h) * W + w] : 0.0f;
  dst[(size_t)((n * H + h) * W + w) * Cst + coff + cc] = (_Float16)v;
}

// ---------------------------------------------------------------------------
// Implicit-GEMM conv, NHWC f16 in, v_wmma_f32_16x16x32_f16 core.
// Block = 64 threads = 2 waves sharing one 32-pixel N tile through LDS:
//   wave w computes Cout tile [blockIdx.y*32 + w*16, +16) over 32 pixels
//   (two 16x16 WMMA tiles per wave, shared A fragment per tile pair).
// Per K-chunk (32 channels of one filter tap):
//   - 64 threads cooperatively stage the 32px x 32K f16 B tile (2 KB) to LDS
//     (zero for spatially OOB taps); both waves read fragments via ds_load.
//   - A fragments come from packed L2-resident weights (2x b128 per lane),
//     with global_prefetch of the next chunk.
//
// Fragment layouts used here (wave32; A and B share contiguous k order):
//  A 16x32 f16 : lane -> M = lane&15; half = lane>>4
//                elems [0..7]  = K (half*16  .. +7)
//                elems [8..15] = K (half*16+8.. +7)
//  B 32x16 f16 : lane -> N = lane&15; elems [0..15] = K (half*16 .. +15)
//  C/D 16x16 f32: lane -> N = lane&15 ; VGPR r -> M = half*8 + r
// ---------------------------------------------------------------------------
__global__ __launch_bounds__(64)
void conv_wmma_kernel(const _Float16* __restrict__ xin, int Cst, int Cin,
                      const _Float16* __restrict__ Wp, int Cpad,
                      const float* __restrict__ bias,
                      float* __restrict__ out32,
                      _Float16* __restrict__ out16a, int coffa, int Csta,
                      _Float16* __restrict__ out16b, int coffb, int Cstb,
                      int IH, int IW, int OH, int OW,
                      int hw_shift, int ow_shift,
                      int KH, int KW, int stride, int pad,
                      int Cout, int act) {
  __shared__ __attribute__((aligned(16))) _Float16 lb[32 * 32];  // 32 px x 32 K

  const int tid  = threadIdx.x;
  const int lane = tid & 31;
  const int wave = tid >> 5;         // 0..1
  const int nq   = lane & 15;
  const int half = lane >> 4;

  // staging mapping: thread -> (pixel, 16-half run)
  const int spix = tid >> 1;
  const int sq   = tid & 1;
  const int sp   = blockIdx.x * 32 + spix;
  const int sn   = sp >> hw_shift;
  const int sr   = sp & ((1 << hw_shift) - 1);
  const int soh  = sr >> ow_shift;
  const int sow  = sr & ((1 << ow_shift) - 1);

  // compute mapping: per-wave Cout tile, two pixels per lane position
  const int co0 = blockIdx.y * 32 + wave * 16;
  const int p0  = blockIdx.x * 32 + nq;
  const int p1  = p0 + 16;
  const int n0 = p0 >> hw_shift, r0 = p0 & ((1 << hw_shift) - 1);
  const int n1 = p1 >> hw_shift, r1 = p1 & ((1 << hw_shift) - 1);
  const int oh0 = r0 >> ow_shift, ow0 = r0 & ((1 << ow_shift) - 1);
  const int oh1 = r1 >> ow_shift, ow1 = r1 & ((1 << ow_shift) - 1);

  const _Float16* wrow = Wp + (size_t)(co0 + nq) * (KH * KW * Cpad);

  v8f acc0 = {}, acc1 = {};

  for (int kh = 0; kh < KH; ++kh) {
    const int sih = soh * stride - pad + kh;
    for (int kw = 0; kw < KW; ++kw) {
      const int siw  = sow * stride - pad + kw;
      const bool sinb = (sih >= 0) && (sih < IH) && (siw >= 0) && (siw < IW);
      const _Float16* sbase = xin + (size_t)((sn * IH + sih) * IW + siw) * Cst;
      const _Float16* ap    = wrow + (kh * KW + kw) * Cpad;
      for (int c0 = 0; c0 < Cin; c0 += 32) {
        __syncthreads();  // previous chunk's LDS reads complete
        // ---- cooperative B staging: this thread's 32B of the 2KB tile ----
        h8 z = {};
        h8 s0 = z, s1 = z;
        if (sinb) {
          s0 = *(const h8*)(sbase + c0 + sq * 16);
          s1 = *(const h8*)(sbase + c0 + sq * 16 + 8);
        }
        *(h8*)(lb + spix * 32 + sq * 16)     = s0;
        *(h8*)(lb + spix * 32 + sq * 16 + 8) = s1;
        // ---- A fragment from packed weights + prefetch next chunk ----
        h8 a0 = *(const h8*)(ap + c0 + half * 16);
        h8 a1 = *(const h8*)(ap + c0 + half * 16 + 8);
        __builtin_prefetch((const void*)(ap + c0 + 32 + half * 8), 0, 1);
        v16h A = __builtin_shufflevector(a0, a1, 0, 1, 2, 3, 4, 5, 6, 7,
                                                 8, 9, 10, 11, 12, 13, 14, 15);
        __syncthreads();  // B tile visible
        // ---- B fragments from LDS ----
        h8 b0l = *(const h8*)(lb + nq * 32 + half * 16);
        h8 b0h = *(const h8*)(lb + nq * 32 + half * 16 + 8);
        h8 b1l = *(const h8*)(lb + (nq + 16) * 32 + half * 16);
        h8 b1h = *(const h8*)(lb + (nq + 16) * 32 + half * 16 + 8);
        v16h B0 = __builtin_shufflevector(b0l, b0h, 0, 1, 2, 3, 4, 5, 6, 7,
                                                    8, 9, 10, 11, 12, 13, 14, 15);
        v16h B1 = __builtin_shufflevector(b1l, b1h, 0, 1, 2, 3, 4, 5, 6, 7,
                                                    8, 9, 10, 11, 12, 13, 14, 15);
        acc0 = __builtin_amdgcn_wmma_f32_16x16x32_f16(false, A, false, B0,
                                                      (short)0, acc0, false, false);
        acc1 = __builtin_amdgcn_wmma_f32_16x16x32_f16(false, A, false, B1,
                                                      (short)0, acc1, false, false);
      }
    }
  }

  // epilogue: bias + activation, store f32 NCHW and/or f16 NHWC slices
#pragma unroll
  for (int r = 0; r < 8; ++r) {
    const int c = co0 + (half << 3) + r;
    if (c < Cout) {
      const float bi = bias[c];
      float v0 = acc0[r] + bi, v1 = acc1[r] + bi;
      if (act == ACT_LEAKY) {
        v0 = (v0 >= 0.0f) ? v0 : 0.2f * v0;
        v1 = (v1 >= 0.0f) ? v1 : 0.2f * v1;
      } else if (act == ACT_RELU) {
        v0 = fmaxf(v0, 0.0f); v1 = fmaxf(v1, 0.0f);
      } else if (act == ACT_SIGMOID) {
        v0 = 1.0f / (1.0f + __expf(-v0)); v1 = 1.0f / (1.0f + __expf(-v1));
      }
      if (out32) {
        out32[((n0 * Cout + c) * OH + oh0) * OW + ow0] = v0;
        out32[((n1 * Cout + c) * OH + oh1) * OW + ow1] = v1;
      }
      if (out16a) {
        out16a[(size_t)((n0 * OH + oh0) * OW + ow0) * Csta + coffa + c] = (_Float16)v0;
        out16a[(size_t)((n1 * OH + oh1) * OW + ow1) * Csta + coffa + c] = (_Float16)v1;
      }
      if (out16b) {
        out16b[(size_t)((n0 * OH + oh0) * OW + ow0) * Cstb + coffb + c] = (_Float16)v0;
        out16b[(size_t)((n1 * OH + oh1) * OW + ow1) * Cstb + coffb + c] = (_Float16)v1;
      }
    }
  }
}

// ---------------------------------------------------------------------------
// Bilinear warp with zero-outside sampling + zw mask; writes f16 directly
// into the NHWC concat buffer at channel CD + c*LK + l.
// ---------------------------------------------------------------------------
__device__ __forceinline__ float bcorner(const float* __restrict__ img,
                                         int xc, int yc, float wgt, int H, int W) {
  if (xc < 0 || xc >= W || yc < 0 || yc >= H) return 0.0f;
  return img[yc * W + xc] * wgt;
}

__global__ void warp_sample_kernel(const float* __restrict__ prev,
                                   const float* __restrict__ d2,
                                   _Float16* __restrict__ cat2, int C2, int CD,
                                   int Bn, int CR, int LK, int H, int W,
                                   int hw_shift, int w_shift) {
  int idx = blockIdx.x * blockDim.x + threadIdx.x;
  const int hw = H * W;
  const int total = Bn * CR * LK * hw;
  if (idx >= total) return;
  const int pix = idx & (hw - 1);
  int t = idx >> hw_shift;
  const int l = t % LK; t /= LK;
  const int c = t % CR; t /= CR;
  const int n = t;
  const int h = pix >> w_shift;
  const int w = pix & (W - 1);
  _Float16* dst = cat2 + (size_t)((n * H + h) * W + w) * C2 + CD + c * LK + l;
  if (l == LK - 1) { *dst = (_Float16)0.0f; return; }  // zw mask
  const float ix = d2[((n * (2 * LK) + l) * H + h) * W + w];
  const float iy = d2[((n * (2 * LK) + LK + l) * H + h) * W + w];
  const float x0 = floorf(ix), y0 = floorf(iy);
  const float wx = ix - x0,    wy = iy - y0;
  const int xi = (int)x0, yi = (int)y0;
  const float* img = prev + (n * CR + c) * hw;
  float v = bcorner(img, xi,     yi,     (1.0f - wx) * (1.0f - wy), H, W)
          + bcorner(img, xi + 1, yi,     wx * (1.0f - wy),          H, W)
          + bcorner(img, xi,     yi + 1, (1.0f - wx) * wy,          H, W)
          + bcorner(img, xi + 1, yi + 1, wx * wy,                   H, W);
  *dst = (_Float16)v;
}

// gated = M * tile(reset): in place on the f16 NHWC concat buffer
__global__ void gate_mul_kernel(_Float16* __restrict__ cat2,
                                const float* __restrict__ reset,
                                int C2, int CD, int Bn, int CR, int LK,
                                int H, int W, int hw_shift, int w_shift) {
  int idx = blockIdx.x * blockDim.x + threadIdx.x;
  const int hw = H * W;
  const int total = Bn * CR * LK * hw;
  if (idx >= total) return;
  const int pix = idx & (hw - 1);
  int t = idx >> hw_shift;
  const int ch = t % (CR * LK);
  const int n  = t / (CR * LK);
  const int h = pix >> w_shift, w = pix & (W - 1);
  _Float16* p = cat2 + (size_t)((n * H + h) * W + w) * C2 + CD + ch;
  const float v = (float)*p * reset[(n * CR + (ch % CR)) * hw + pix];
  *p = (_Float16)v;
}

// h = prev*u + o*(1-u)
__global__ void gru_combine_kernel(const float* __restrict__ prev,
                                   const float* __restrict__ upd,
                                   const float* __restrict__ o,
                                   float* __restrict__ hout, int total) {
  int idx = blockIdx.x * blockDim.x + threadIdx.x;
  if (idx >= total) return;
  const float u = upd[idx];
  hout[idx] = prev[idx] * u + o[idx] * (1.0f - u);
}

// ---------------------------------------------------------------------------
// Host orchestration
// ---------------------------------------------------------------------------
extern "C" void kernel_launch(void* const* d_in, const int* in_sizes, int n_in,
                              void* d_out, int out_size, void* d_ws, size_t ws_size,
                              hipStream_t stream) {
  (void)in_sizes; (void)n_in; (void)out_size; (void)ws_size;
  const int Bn = 4;
  const float* x = (const float*)d_in[0];
  const float* hprev[3] = {(const float*)d_in[1], (const float*)d_in[2], (const float*)d_in[3]};
  const int CDv[3] = {32, 96, 96};
  const int CRv[3] = {64, 96, 96};
  const int LKv[3] = {13, 13, 9};
  const int SZv[3] = {64, 32, 16};

  // bump allocator over d_ws (256B aligned, slack after each buffer)
  char* base = (char*)d_ws;
  size_t off = 0;
  auto alloc = [&](size_t bytes) -> void* {
    void* p = base + off;
    off += (bytes + 511) & ~(size_t)255;
    return p;
  };
  _Float16* xin  = (_Float16*)alloc(1048576 * 2);   // NHWC input, max 4*64*64*64
  _Float16* cat1 = (_Float16*)alloc(1572864 * 2);   // concat(xds,h)   max C=96 @64^2
  _Float16* d1p  = (_Float16*)alloc(524288 * 2);    // d1, C=32 @64^2
  float*    d2b  = (float*)   alloc(425984 * 4);    // flow fields f32 NCHW
  _Float16* cat2 = (_Float16*)alloc(14155776 * 2);  // concat(xds,M)  max C=864 @64^2
  float*    rb_  = (float*)   alloc(1048576 * 4);
  float*    ub_  = (float*)   alloc(1048576 * 4);
  float*    ob_  = (float*)   alloc(1048576 * 4);
  _Float16* wpk  = (_Float16*)alloc(1179648 * 2);   // packed weights, max 96*9*1344

  float* outp = (float*)d_out;
  float* houts[3] = {outp, outp + 1048576, outp + 1048576 + 393216};

  auto packw = [&](const float* src, int Cout, int Cin, int KH, int KW,
                   int& Coutp, int& Cpad) {
    Coutp = (Cout + 31) & ~31;   // two 16-row wave tiles per block
    Cpad  = (Cin + 31) & ~31;
    const int tot = Coutp * KH * KW * Cpad;
    pack_w_kernel<<<(tot + 255) / 256, 256, 0, stream>>>(src, wpk, Cout, Cin, KH, KW, Coutp, Cpad);
  };
  auto packa = [&](const float* src, _Float16* dst, int C, int H, int W,
                   int Cst, int coff, int Cfill) {
    const int tot = Bn * H * W * Cfill;
    pack_a_kernel<<<(tot + 255) / 256, 256, 0, stream>>>(src, dst, Bn, C, H, W, Cst, coff, Cfill);
  };
  auto conv = [&](const _Float16* in, int Cst, int Cin, const float* Wsrc,
                  const float* bi, float* o32,
                  _Float16* o16a, int coffa, int Csta,
                  _Float16* o16b, int coffb, int Cstb,
                  int IH, int IW, int OH, int OW, int hw_sh, int ow_sh,
                  int KH, int KW, int st, int pd, int Cout, int act) {
    int Coutp, Cpad;
    packw(Wsrc, Cout, Cin, KH, KW, Coutp, Cpad);
    dim3 grid((unsigned)((Bn * OH * OW) / 32), (unsigned)(Coutp / 32));
    conv_wmma_kernel<<<grid, dim3(64), 0, stream>>>(
        in, Cst, Cin, wpk, Cpad, bi, o32, o16a, coffa, Csta, o16b, coffb, Cstb,
        IH, IW, OH, OW, hw_sh, ow_sh, KH, KW, st, pd, Cout, act);
  };

  // layer0 input: x (C=1) packed to NHWC with Cst=8 (zero-padded channels)
  packa(x, xin, 1, 128, 128, 8, 0, 8);
  int inC = 1, inCst = 8, inH = 128;

  const int hwsh[3] = {12, 10, 8};  // log2(S*S)
  const int owsh[3] = {6, 5, 4};    // log2(S)

  for (int i = 0; i < 3; ++i) {
    const int base_i = 4 + 12 * i;  // params flattened in dict order per layer
    const float* dsw = (const float*)d_in[base_i + 0];
    const float* dsb = (const float*)d_in[base_i + 1];
    const float* dw1 = (const float*)d_in[base_i + 2];
    const float* db1 = (const float*)d_in[base_i + 3];
    const float* dw2 = (const float*)d_in[base_i + 4];
    const float* db2 = (const float*)d_in[base_i + 5];
    const float* rw  = (const float*)d_in[base_i + 6];
    const float* rbi = (const float*)d_in[base_i + 7];
    const float* uw  = (const float*)d_in[base_i + 8];
    const float* ubi = (const float*)d_in[base_i + 9];
    const float* owt = (const float*)d_in[base_i + 10];
    const float* obi = (const float*)d_in[base_i + 11];
    const int CD = CDv[i], CR = CRv[i], LK = LKv[i], S = SZv[i];
    const int C1 = CD + CR, C2 = CD + CR * LK;

    // downsample 4x4 s2 p1, leaky -> f16 NHWC into both concat buffers (offset 0)
    conv(xin, inCst, inC, dsw, dsb, nullptr,
         cat1, 0, C1, cat2, 0, C2,
         inH, inH, S, S, hwsh[i], owsh[i], 4, 4, 2, 1, CD, ACT_LEAKY);
    // h -> cat1 at channel offset CD
    packa(hprev[i], cat1, CR, S, S, C1, CD, CR);
    // d1 = leaky(conv5x5(cat1)) -> f16 NHWC (C=32)
    conv(cat1, C1, C1, dw1, db1, nullptr, d1p, 0, 32, nullptr, 0, 0,
         S, S, S, S, hwsh[i], owsh[i], 5, 5, 1, 2, 32, ACT_LEAKY);
    // d2 = relu(conv5x5(d1)+b) -> f32 NCHW (flow fields)
    conv(d1p, 32, 32, dw2, db2, d2b, nullptr, 0, 0, nullptr, 0, 0,
         S, S, S, S, hwsh[i], owsh[i], 5, 5, 1, 2, 2 * LK, ACT_RELU);
    // warp: bilinear sample prev with flows -> cat2 channels [CD, CD+CR*LK)
    const int totM = Bn * CR * LK * S * S;
    warp_sample_kernel<<<(totM + 255) / 256, 256, 0, stream>>>(
        hprev[i], d2b, cat2, C2, CD, Bn, CR, LK, S, S, hwsh[i], owsh[i]);
    // reset / update: 3x3 conv over cat2, sigmoid -> f32 NCHW
    conv(cat2, C2, C2, rw, rbi, rb_, nullptr, 0, 0, nullptr, 0, 0,
         S, S, S, S, hwsh[i], owsh[i], 3, 3, 1, 1, CR, ACT_SIGMOID);
    conv(cat2, C2, C2, uw, ubi, ub_, nullptr, 0, 0, nullptr, 0, 0,
         S, S, S, S, hwsh[i], owsh[i], 3, 3, 1, 1, CR, ACT_SIGMOID);
    // gated = M * tile(reset) in place in cat2
    gate_mul_kernel<<<(totM + 255) / 256, 256, 0, stream>>>(
        cat2, rb_, C2, CD, Bn, CR, LK, S, S, hwsh[i], owsh[i]);
    // o = leaky(conv3x3(cat2))
    conv(cat2, C2, C2, owt, obi, ob_, nullptr, 0, 0, nullptr, 0, 0,
         S, S, S, S, hwsh[i], owsh[i], 3, 3, 1, 1, CR, ACT_LEAKY);
    // h' = prev*u + o*(1-u) -> d_out
    const int totH = Bn * CR * S * S;
    gru_combine_kernel<<<(totH + 255) / 256, 256, 0, stream>>>(
        hprev[i], ub_, ob_, houts[i], totH);

    if (i < 2) {
      // next layer input: h' packed NHWC f16
      packa(houts[i], xin, CR, S, S, CR, 0, CR);
      inC = CR; inCst = CR; inH = S;
    }
  }
}
